// SpatioTemporalTransformer_83176336655140
// MI455X (gfx1250) — matrix operand order repaired and checked by reading, not compiled, not run on hardware
//
#include <hip/hip_runtime.h>
#include <math.h>
#include <stdint.h>

// ---------------- CDNA5 types ----------------
typedef __attribute__((ext_vector_type(16))) __bf16 v16bf;
typedef __attribute__((ext_vector_type(8)))  float  v8f;

struct alignas(16) B128 { unsigned x, y, z, w; };
struct alignas(32) B256 { B128 lo, hi; };
struct alignas(16) F4   { float x, y, z, w; };
struct alignas(8)  F2   { float x, y; };

// ---------------- model constants ----------------
#define S_TOT 4112      // T*G
#define G_LEN 257
#define T_LEN 16
#define D_MOD 512
#define H_NUM 8
#define B_SZ  4
#define ROWS  (B_SZ * S_TOT)   // 16448 real token rows
#define M_PAD 16512            // 129*128: padded GEMM M (pad rows never consumed)
#define TOKN  256
#define V_SZ  1024

__device__ __forceinline__ float wredMax(float v) {
    for (int o = 16; o; o >>= 1) v = fmaxf(v, __shfl_xor(v, o, 32));
    return v;
}
__device__ __forceinline__ float wredSum(float v) {
    for (int o = 16; o; o >>= 1) v += __shfl_xor(v, o, 32);
    return v;
}

// =====================================================================
// Embedding: h[b,s,d] = tok_emb + pos_emb   (f32)
// =====================================================================
__global__ void embed_k(const int* __restrict__ x, const float* __restrict__ img,
                        const float* __restrict__ act, const float* __restrict__ spe,
                        const float* __restrict__ tmp, float* __restrict__ h)
{
    size_t idx = (size_t)blockIdx.x * blockDim.x + threadIdx.x;   // exact grid
    int d   = (int)(idx & 511);
    int s   = (int)((idx >> 9) % S_TOT);
    int b   = (int)((idx >> 9) / S_TOT);
    int pig = s % G_LEN;
    int grp = s / G_LEN;
    int xv  = x[(size_t)b * S_TOT + s];
    float te;
    if (pig == 0) {
        int a = xv < 17 ? xv : 17;
        te = act[a * D_MOD + d];
    } else {
        te = img[(size_t)xv * D_MOD + d];
    }
    float pe = tmp[grp * D_MOD + d] + (pig == 0 ? 0.f : spe[(pig - 1) * D_MOD + d]);
    h[idx] = te + pe;
}

// =====================================================================
// LayerNorm over D=512, one row per block (128 thr x float4).
// Writes bf16 (GEMM input). If wb != null also writes normalized f32
// back (needed for the FFN residual which is off the normalized value).
// =====================================================================
__global__ __launch_bounds__(128)
void ln_k(const float* __restrict__ in, const float* __restrict__ sc,
          const float* __restrict__ bi, __bf16* __restrict__ out,
          float* __restrict__ wb)
{
    __shared__ float red[128];
    int row = blockIdx.x, tid = threadIdx.x;
    const F4* xr = (const F4*)(in + (size_t)row * D_MOD);
    F4 v = xr[tid];
    red[tid] = v.x + v.y + v.z + v.w;
    __syncthreads();
    for (int s2 = 64; s2 > 0; s2 >>= 1) { if (tid < s2) red[tid] += red[tid + s2]; __syncthreads(); }
    float mean = red[0] * (1.f / 512.f);
    __syncthreads();
    float dx = v.x - mean, dy = v.y - mean, dz = v.z - mean, dw = v.w - mean;
    red[tid] = dx*dx + dy*dy + dz*dz + dw*dw;
    __syncthreads();
    for (int s2 = 64; s2 > 0; s2 >>= 1) { if (tid < s2) red[tid] += red[tid + s2]; __syncthreads(); }
    float rstd = rsqrtf(red[0] * (1.f / 512.f) + 1e-5f);
    int d = tid * 4;
    float o0 = dx * rstd * sc[d]     + bi[d];
    float o1 = dy * rstd * sc[d + 1] + bi[d + 1];
    float o2 = dz * rstd * sc[d + 2] + bi[d + 2];
    float o3 = dw * rstd * sc[d + 3] + bi[d + 3];
    __bf16* op = out + (size_t)row * D_MOD + d;
    op[0] = (__bf16)o0; op[1] = (__bf16)o1; op[2] = (__bf16)o2; op[3] = (__bf16)o3;
    if (wb) {
        F4 o; o.x = o0; o.y = o1; o.z = o2; o.w = o3;
        ((F4*)(wb + (size_t)row * D_MOD))[tid] = o;
    }
}

// =====================================================================
// Weight convert+transpose: W (KxN f32, row-major) -> Wt (NxK bf16).
// =====================================================================
__global__ __launch_bounds__(256)
void transp_k(const float* __restrict__ W, __bf16* __restrict__ Wt, int K, int N)
{
    __shared__ float tile[32][33];
    int nx = blockIdx.x * 32 + threadIdx.x;    // N index
    int k0 = blockIdx.y * 32;                  // K base
    for (int j = 0; j < 32; j += 8)
        tile[threadIdx.y + j][threadIdx.x] = W[(size_t)(k0 + threadIdx.y + j) * N + nx];
    __syncthreads();
    int kx = k0 + threadIdx.x;
    int n1 = blockIdx.x * 32;
    for (int j = 0; j < 32; j += 8)
        Wt[(size_t)(n1 + threadIdx.y + j) * K + kx] = (__bf16)tile[threadIdx.x][threadIdx.y + j];
}

// =====================================================================
// bf16 WMMA GEMM:  C[M,N] = A[M,K] @ Wt[N,K]^T (+bias)
// 128x128x32 block tile, 8 waves (2x4), wave = 64x32 patch -> 8 WMMAs
// per K-step. Double-buffered LDS fed by CDNA5
// global_load_async_to_lds_b128 (ASYNCcnt), one wait+barrier per step.
// M must be a multiple of 128 (callers pad).
// MODE 0: Cf = v | MODE 1: Cf += v | MODE 2: Cb = gelu(v) bf16
// =====================================================================
template<int MODE>
__global__ __launch_bounds__(256)
void gemm_bf16_k(const __bf16* __restrict__ A, const __bf16* __restrict__ Wt,
                 const float* __restrict__ bias, float* __restrict__ Cf,
                 __bf16* __restrict__ Cb, int M, int N, int K)
{
    __shared__ __bf16 As[2][128][40];   // pad 32->40 to dodge bank conflicts
    __shared__ __bf16 Bs[2][128][40];   // Bs[buf][n][k]
    const int tid  = threadIdx.x;
    const int lane = tid & 31;
    const int wave = tid >> 5;
    const int m0 = blockIdx.x * 128;
    const int n0 = blockIdx.y * 128;
    const int wm = (wave >> 2) * 64;     // 0 / 64
    const int wn = (wave & 3) * 32;      // 0..96

    v8f acc[4][2] = {};

    const int lr = tid >> 1;             // 0..127 tile row
    const int lk = (tid & 1) * 16;       // 0 / 16
    const __bf16* aG = A  + (size_t)(m0 + lr) * K + lk;   // rows always valid (padded)
    const __bf16* bG = Wt + (size_t)(n0 + lr) * K + lk;
    // raw LDS byte offsets for the async-to-LDS destination (flat addr low 32b)
    const unsigned aL[2] = { (unsigned)(uintptr_t)&As[0][lr][lk],
                             (unsigned)(uintptr_t)&As[1][lr][lk] };
    const unsigned bL[2] = { (unsigned)(uintptr_t)&Bs[0][lr][lk],
                             (unsigned)(uintptr_t)&Bs[1][lr][lk] };

    const int am = lane & 15, ak = (lane >> 4) * 8;   // A frag lane map (ISA 7.12.2)
    const int bn = lane & 15, bk = (lane >> 4) * 8;   // B frag lane map

    // ---- prologue: async-stage tile 0 into buffer 0 ----
    asm volatile(
        "global_load_async_to_lds_b128 %0, %2, off\n\t"
        "global_load_async_to_lds_b128 %0, %2, off offset:16\n\t"
        "global_load_async_to_lds_b128 %1, %3, off\n\t"
        "global_load_async_to_lds_b128 %1, %3, off offset:16"
        :: "v"(aL[0]), "v"(bL[0]), "v"(aG), "v"(bG) : "memory");
    asm volatile("s_wait_asynccnt 0" ::: "memory");
    __syncthreads();

    const int nk = K >> 5;
    for (int kk = 0; kk < nk; ++kk) {
        const int cur = kk & 1;
        if (kk + 1 < nk) {
            const __bf16* aN = aG + (size_t)(kk + 1) * 32;
            const __bf16* bN = bG + (size_t)(kk + 1) * 32;
            asm volatile(
                "global_load_async_to_lds_b128 %0, %2, off\n\t"
                "global_load_async_to_lds_b128 %0, %2, off offset:16\n\t"
                "global_load_async_to_lds_b128 %1, %3, off\n\t"
                "global_load_async_to_lds_b128 %1, %3, off offset:16"
                :: "v"(cur ? aL[0] : aL[1]), "v"(cur ? bL[0] : bL[1]),
                   "v"(aN), "v"(bN) : "memory");
        }
        v16bf af[4], bfr[2];
        for (int i = 0; i < 4; ++i) {
            B256 t;
            t.lo = *(const B128*)&As[cur][wm + i * 16 + am][ak];        // K: ak..ak+7
            t.hi = *(const B128*)&As[cur][wm + i * 16 + am][ak + 16];   // K: ak+16..+23
            af[i] = __builtin_bit_cast(v16bf, t);
        }
        for (int j = 0; j < 2; ++j) {
            B256 t;
            t.lo = *(const B128*)&Bs[cur][wn + j * 16 + bn][bk];        // K: bk..bk+7
            t.hi = *(const B128*)&Bs[cur][wn + j * 16 + bn][bk + 16];   // K: bk+16..+23
            bfr[j] = __builtin_bit_cast(v16bf, t);
        }
        for (int i = 0; i < 4; ++i)
            for (int j = 0; j < 2; ++j)
                acc[i][j] = __builtin_amdgcn_wmma_f32_16x16x32_bf16(
                    false, af[i], false, bfr[j], (short)0, acc[i][j], false, false);
        if (kk + 1 < nk)
            asm volatile("s_wait_asynccnt 0" ::: "memory");
        __syncthreads();
    }

    // ---- epilogue ----
    const int cn = lane & 15;
    const int cm = (lane >> 4) * 8;
    for (int j = 0; j < 2; ++j) {
        int col = n0 + wn + j * 16 + cn;
        float bv = bias ? bias[col] : 0.f;
        for (int i = 0; i < 4; ++i) {
            for (int r = 0; r < 8; ++r) {
                int row = m0 + wm + i * 16 + cm + r;
                float vv = acc[i][j][r] + bv;
                size_t o = (size_t)row * N + col;
                if (MODE == 0) {
                    Cf[o] = vv;
                } else if (MODE == 1) {
                    Cf[o] += vv;
                } else {
                    float g = 0.5f * vv * (1.f + tanhf(0.7978845608028654f *
                                  (vv + 0.044715f * vv * vv * vv)));
                    Cb[o] = (__bf16)g;
                }
            }
        }
    }
}

// =====================================================================
// LDS-staged attention. One 256-thr workgroup per (group, head):
// stages the group's K (row stride 65, bank-conflict-free) and V
// (row stride 64, read as aligned float2) into dynamic LDS ONCE, then
// 8 waves sweep queries q = wave, wave+8, ... with no block barriers
// inside the query loop (wave-internal LDS is retired in order).
// Global traffic per layer drops from ~17 GB (per-query K/V re-reads)
// to the ~0.07 GB footprint of qkv itself.
// Dynamic LDS: Vs[seq*64] | Ks[seq*65] | pb[8*seq] | qb[8*64]
//   seq=257 -> 142,884 B ; seq=16 -> 10,816 B  (<= 320 KB/WGP)
// =====================================================================
__global__ __launch_bounds__(256)
void attn_k(const float* __restrict__ qkv, __bf16* __restrict__ out,
            int seqlen, int causal, int temporal)
{
    extern __shared__ float smem[];
    float* Vs = smem;                         // seqlen*64 (16B-aligned base)
    float* Ks = Vs + seqlen * 64;             // seqlen*65
    float* pb = Ks + seqlen * 65;             // 8*seqlen
    float* qb = pb + 8 * seqlen;              // 8*64

    const int tid = threadIdx.x, lane = tid & 31, w = tid >> 5;
    const int gid  = blockIdx.x;
    const int head = gid & 7;
    const int grp  = gid >> 3;
    int base, stride;
    if (temporal) { base = (grp / G_LEN) * S_TOT + (grp % G_LEN); stride = G_LEN; }
    else          { base = grp * G_LEN;                           stride = 1; }

    // ---- stage K/V for this (group, head) ----
    const int nel = seqlen * 64;
    for (int idx = tid; idx < nel; idx += 256) {
        int r = idx >> 6, d = idx & 63;
        size_t t = (size_t)(base + r * stride) * 1536 + head * 64 + d;
        Ks[r * 65 + d] = qkv[t + 512];
        Vs[r * 64 + d] = qkv[t + 1024];
    }
    __syncthreads();

    float* pw = pb + w * seqlen;
    for (int q = w; q < seqlen; q += 8) {
        const size_t qtok = (size_t)(base + q * stride);
        const float* qp = qkv + qtok * 1536 + head * 64;
        qb[w * 64 + 2 * lane]     = qp[2 * lane];
        qb[w * 64 + 2 * lane + 1] = qp[2 * lane + 1];
        __builtin_amdgcn_wave_barrier();

        const int kmax = causal ? (q + 1) : seqlen;
        // pass 1: raw scores -> pw, lane max
        float mloc = -__builtin_inff();
        for (int j = lane; j < seqlen; j += 32) {
            float s = -__builtin_inff();
            if (j < kmax) {
                float d = 0.f;
                for (int e = 0; e < 64; ++e)
                    d = fmaf(qb[w * 64 + e], Ks[j * 65 + e], d);
                s = d * 0.125f;                  // 1/sqrt(64)
            }
            pw[j] = s;
            mloc = fmaxf(mloc, s);
        }
        float m = wredMax(mloc);
        __builtin_amdgcn_wave_barrier();
        // pass 2: exponentiate in place, lane sum (masked keys -> exactly 0)
        float lloc = 0.f;
        for (int j = lane; j < seqlen; j += 32) {
            float p = __expf(pw[j] - m);
            pw[j] = p;
            lloc += p;
        }
        float l = wredSum(lloc);
        __builtin_amdgcn_wave_barrier();
        // PV: broadcast p, float2 V rows, branchless
        float a0 = 0.f, a1 = 0.f;
        for (int j = 0; j < seqlen; ++j) {
            float pj = pw[j];
            F2 vv = ((const F2*)(Vs + j * 64))[lane];
            a0 = fmaf(pj, vv.x, a0);
            a1 = fmaf(pj, vv.y, a1);
        }
        float inv = 1.f / l;
        __bf16* op = out + qtok * D_MOD + head * 64;
        op[2 * lane]     = (__bf16)(a0 * inv);
        op[2 * lane + 1] = (__bf16)(a1 * inv);
        __builtin_amdgcn_wave_barrier();
    }
}

// =====================================================================
// Gather last TOK rows of each batch into bf16 for the head GEMM.
// =====================================================================
__global__ void gather_head_k(const float* __restrict__ h, __bf16* __restrict__ o)
{
    int idx = blockIdx.x * 256 + threadIdx.x;   // exact: 1024*512
    int d = idx & 511, r = idx >> 9;
    int b = r >> 8, t = r & 255;
    size_t tok = (size_t)b * S_TOT + (S_TOT - TOKN) + t;
    o[idx] = (__bf16)h[tok * D_MOD + d];
}

// =====================================================================
// Final mask: out = tok==MASK ? logits : (v==tok ? 1 : -inf)
// =====================================================================
__global__ void mask_k(const int* __restrict__ x, const float* __restrict__ logits,
                       float* __restrict__ out)
{
    int idx = blockIdx.x * 256 + threadIdx.x;   // exact: 1048576
    int v = idx & 1023, r = idx >> 10;
    int t = r & 255, b = r >> 8;
    int tok = x[(size_t)b * S_TOT + (S_TOT - TOKN) + t];
    float res;
    if (tok == 1024)      res = logits[(size_t)r * V_SZ + v];
    else if (v == tok)    res = 1.0f;
    else                  res = -__builtin_inff();
    out[idx] = res;
}

// =====================================================================
// Host launch
// =====================================================================
extern "C" void kernel_launch(void* const* d_in, const int* in_sizes, int n_in,
                              void* d_out, int out_size, void* d_ws, size_t ws_size,
                              hipStream_t stream)
{
    (void)in_sizes; (void)n_in; (void)out_size; (void)ws_size;

    const int*   x       = (const int*)  d_in[0];
    const float* img_emb = (const float*)d_in[1];
    const float* act_emb = (const float*)d_in[2];
    const float* sp_emb  = (const float*)d_in[3];
    const float* tmp_emb = (const float*)d_in[4];
    const float* sp_ln_s = (const float*)d_in[5];
    const float* sp_ln_b = (const float*)d_in[6];
    const float* sp_wqkv = (const float*)d_in[7];
    const float* sp_bqkv = (const float*)d_in[8];
    const float* sp_wo   = (const float*)d_in[9];
    const float* sp_bo   = (const float*)d_in[10];
    const float* tm_ln_s = (const float*)d_in[11];
    const float* tm_ln_b = (const float*)d_in[12];
    const float* tm_wqkv = (const float*)d_in[13];
    const float* tm_bqkv = (const float*)d_in[14];
    const float* tm_wo   = (const float*)d_in[15];
    const float* tm_bo   = (const float*)d_in[16];
    const float* ln_s    = (const float*)d_in[17];
    const float* ln_b    = (const float*)d_in[18];
    const float* ffn_w1  = (const float*)d_in[19];
    const float* ffn_b1  = (const float*)d_in[20];
    const float* ffn_w2  = (const float*)d_in[21];
    const float* ffn_b2  = (const float*)d_in[22];
    const float* head_w  = (const float*)d_in[23];
    const float* head_b  = (const float*)d_in[24];
    float* out = (float*)d_out;

    // ---- workspace bump allocator (~238 MB) ----
    char* ws = (char*)d_ws;
    size_t off = 0;
    auto take = [&](size_t bytes) -> char* {
        char* p = ws + off;
        off += (bytes + 255) & ~(size_t)255;
        return p;
    };
    float*  h      = (float*) take((size_t)M_PAD * D_MOD * 4);
    __bf16* xln    = (__bf16*)take((size_t)M_PAD * D_MOD * 2);
    float*  qkv    = (float*) take((size_t)M_PAD * 1536  * 4);
    __bf16* attn_o = (__bf16*)take((size_t)M_PAD * D_MOD * 2);
    __bf16* wspq   = (__bf16*)take(8ull * 1536 * 512 * 2);
    __bf16* wspo   = (__bf16*)take(8ull * 512  * 512 * 2);
    __bf16* wtmq   = (__bf16*)take(8ull * 1536 * 512 * 2);
    __bf16* wtmo   = (__bf16*)take(8ull * 512  * 512 * 2);
    __bf16* wf1    = (__bf16*)take(8ull * 2048 * 512 * 2);
    __bf16* wf2    = (__bf16*)take(8ull * 512  * 2048 * 2);
    __bf16* whd    = (__bf16*)take(1024ull * 512 * 2);
    __bf16* ffn_mid = (__bf16*)qkv;   // alias: qkv dead during FFN (102MB >= 68MB)
    float*  logits  = qkv;            // alias: qkv dead at head

    // ---- one-time weight convert + transpose (f32 KxN -> bf16 NxK) ----
    dim3 tb(32, 8);
    for (int l = 0; l < 8; ++l) {
        transp_k<<<dim3(1536/32, 512/32),  tb, 0, stream>>>(sp_wqkv + (size_t)l*512*1536, wspq + (size_t)l*1536*512, 512, 1536);
        transp_k<<<dim3(512/32,  512/32),  tb, 0, stream>>>(sp_wo   + (size_t)l*512*512,  wspo + (size_t)l*512*512,  512, 512);
        transp_k<<<dim3(1536/32, 512/32),  tb, 0, stream>>>(tm_wqkv + (size_t)l*512*1536, wtmq + (size_t)l*1536*512, 512, 1536);
        transp_k<<<dim3(512/32,  512/32),  tb, 0, stream>>>(tm_wo   + (size_t)l*512*512,  wtmo + (size_t)l*512*512,  512, 512);
        transp_k<<<dim3(2048/32, 512/32),  tb, 0, stream>>>(ffn_w1  + (size_t)l*512*2048, wf1  + (size_t)l*2048*512, 512, 2048);
        transp_k<<<dim3(512/32,  2048/32), tb, 0, stream>>>(ffn_w2  + (size_t)l*2048*512, wf2  + (size_t)l*512*2048, 2048, 512);
    }
    transp_k<<<dim3(1024/32, 512/32), tb, 0, stream>>>(head_w, whd, 512, 1024);

    // ---- embedding (real rows only; pad rows feed only pad outputs) ----
    embed_k<<<(ROWS * D_MOD) / 256, 256, 0, stream>>>(x, img_emb, act_emb, sp_emb, tmp_emb, h);

    const dim3 gQKV(129, 12), gD(129, 4), gF1(129, 16);
    // dynamic LDS for attention: Vs + Ks + pb + qb (floats)
    const int shSp = (G_LEN*64 + G_LEN*65 + 8*G_LEN + 8*64) * 4;   // 142,884 B
    const int shTm = (T_LEN*64 + T_LEN*65 + 8*T_LEN + 8*64) * 4;   // 10,816 B
    const int spBlocks = (B_SZ * T_LEN) * H_NUM;                   // 512
    const int tmBlocks = (B_SZ * G_LEN) * H_NUM;                   // 8224

    for (int l = 0; l < 8; ++l) {
        // -- spatial attention block --
        ln_k<<<ROWS, 128, 0, stream>>>(h, sp_ln_s + l*512, sp_ln_b + l*512, xln, nullptr);
        gemm_bf16_k<0><<<gQKV, 256, 0, stream>>>(xln, wspq + (size_t)l*1536*512, sp_bqkv + l*1536,
                                                 qkv, nullptr, M_PAD, 1536, 512);
        attn_k<<<spBlocks, 256, shSp, stream>>>(qkv, attn_o, G_LEN, 0, 0);
        gemm_bf16_k<1><<<gD, 256, 0, stream>>>(attn_o, wspo + (size_t)l*512*512, sp_bo + l*512,
                                               h, nullptr, M_PAD, 512, 512);
        // -- temporal (causal) attention block --
        ln_k<<<ROWS, 128, 0, stream>>>(h, tm_ln_s + l*512, tm_ln_b + l*512, xln, nullptr);
        gemm_bf16_k<0><<<gQKV, 256, 0, stream>>>(xln, wtmq + (size_t)l*1536*512, tm_bqkv + l*1536,
                                                 qkv, nullptr, M_PAD, 1536, 512);
        attn_k<<<tmBlocks, 256, shTm, stream>>>(qkv, attn_o, T_LEN, 1, 1);
        gemm_bf16_k<1><<<gD, 256, 0, stream>>>(attn_o, wtmo + (size_t)l*512*512, tm_bo + l*512,
                                               h, nullptr, M_PAD, 512, 512);
        // -- FFN: h = hn + gelu(hn@W1+b1)@W2+b2 (residual off normalized hn) --
        ln_k<<<ROWS, 128, 0, stream>>>(h, ln_s + l*512, ln_b + l*512, xln, h);
        gemm_bf16_k<2><<<gF1, 256, 0, stream>>>(xln, wf1 + (size_t)l*2048*512, ffn_b1 + l*2048,
                                                nullptr, ffn_mid, M_PAD, 2048, 512);
        gemm_bf16_k<1><<<gD, 256, 0, stream>>>(ffn_mid, wf2 + (size_t)l*512*2048, ffn_b2 + l*512,
                                               h, nullptr, M_PAD, 512, 2048);
    }

    // ---- head + mask ----
    gather_head_k<<<(1024 * 512) / 256, 256, 0, stream>>>(h, xln);
    gemm_bf16_k<0><<<dim3(8, 8), 256, 0, stream>>>(xln, whd, head_b, logits, nullptr,
                                                   1024, 1024, 512);
    mask_k<<<(B_SZ * TOKN * V_SZ) / 256, 256, 0, stream>>>(x, logits, out);
}
